// NestedGIN_23888608100576
// MI455X (gfx1250) — matrix operand order repaired and checked by reading, not compiled, not run on hardware
//
#include <hip/hip_runtime.h>
#include <hip/hip_bf16.h>
#include <hip/hip_fp16.h>

typedef __attribute__((ext_vector_type(16))) _Float16 v16h;
typedef __attribute__((ext_vector_type(8)))  float    v8f;

#define HID      128
#define N_NODES  50000
#define N_EDGES  600000
#define N_GRAPHS 512
#define P_TOTAL  600000

// ---------------------------------------------------------------------------
// z_emb[e,:] = sum_{p in segment(e)} zw[pos_index[p],:] * pos_enc[p]
// pos_batch is sorted -> deterministic segmented sum via binary search.
// Output is written directly as f16 in WMMA A-fragment layout:
//   tile = e/16, r = e%16, feature j -> kk = j/32, jin = j%32,
//   hi = (jin>>3)&1, h = (jin&7)|((jin>>4)<<3), lane = hi*16 + r,
//   flat = tile*2048 + kk*512 + lane*16 + h
// so the edge GEMM loads each A operand as one contiguous v16h per K-chunk.
// ---------------------------------------------------------------------------
__global__ __launch_bounds__(128) void k_zemb(const float* __restrict__ zw,
                                              const int* __restrict__ pidx,
                                              const float* __restrict__ penc,
                                              const int* __restrict__ pbatch,
                                              _Float16* __restrict__ zembh) {
  int e = blockIdx.x;
  int j = threadIdx.x;
  int lo = 0, hi = P_TOTAL;
  while (lo < hi) { int mid = (lo + hi) >> 1; if (pbatch[mid] < e) lo = mid + 1; else hi = mid; }
  float acc = 0.f;
  for (int p = lo; p < P_TOTAL && pbatch[p] == e; ++p)
    acc += zw[pidx[p] * HID + j] * penc[p];

  int kk   = j >> 5;
  int jin  = j & 31;
  int fhi  = (jin >> 3) & 1;
  int fh   = (jin & 7) | ((jin >> 4) << 3);
  int lane = fhi * 16 + (e & 15);
  size_t idx = (size_t)(e >> 4) * 2048 + kk * 512 + lane * 16 + fh;
  zembh[idx] = (_Float16)acc;
}

// ---------------------------------------------------------------------------
// Pack a row-major f32 [128 (K) x 128 (N)] weight into f16 B-fragment layout:
// out[((n16*4 + kk)*32 + lane)*16 + h] = W[(kk*32 + (lane>>4)*8 + (h&7) + (h>>3)*16)*128 + n16*16 + (lane&15)]
// ---------------------------------------------------------------------------
__global__ __launch_bounds__(256) void k_pack_w(const float* __restrict__ W,
                                                _Float16* __restrict__ out) {
  int i    = blockIdx.x * 256 + threadIdx.x;   // 0..16383
  int h    = i & 15;
  int lane = (i >> 4) & 31;
  int kk   = (i >> 9) & 3;
  int n16  = i >> 11;
  int K = kk * 32 + (lane >> 4) * 8 + (h & 7) + (h >> 3) * 16;
  int N = n16 * 16 + (lane & 15);
  out[i] = (_Float16)W[K * HID + N];
}

// A fragment built from row-major f32 activations (16-bit A 16x32 ISA layout):
// lane<16: halves 0..7 -> K=kk*32+0..7,  halves 8..15 -> K=kk*32+16..23
// lane>=16: halves 0..7 -> K=kk*32+8..15, halves 8..15 -> K=kk*32+24..31
__device__ inline v16h load_a_frag(const float* __restrict__ A0,
                                   const float* __restrict__ A1,
                                   int row, int kk, int hi) {
  const float* b0 = A0 + (size_t)row * HID + kk * 32 + hi * 8;
  float f[16];
#pragma unroll
  for (int t = 0; t < 8; ++t) f[t]     = b0[t];
#pragma unroll
  for (int t = 0; t < 8; ++t) f[8 + t] = b0[16 + t];
  if (A1) {
    const float* b1 = A1 + (size_t)row * HID + kk * 32 + hi * 8;
#pragma unroll
    for (int t = 0; t < 8; ++t) f[t]     += b1[t];
#pragma unroll
    for (int t = 0; t < 8; ++t) f[8 + t] += b1[16 + t];
  }
  v16h a;
#pragma unroll
  for (int t = 0; t < 16; ++t) a[t] = (_Float16)f[t];
  return a;
}

// ---------------------------------------------------------------------------
// Fused edge layer: ea = z_emb @ We + be  (WMMA, f32 accum, A pre-packed f16),
// msg = relu(x[src] + ea), atomic scatter-add into agg[dst].
// Block = 256 threads = 8 waves; block tile = 16 edges x 128 cols.
// ---------------------------------------------------------------------------
__global__ __launch_bounds__(256) void k_edge_gine(const _Float16* __restrict__ zembh,
                                                   const _Float16* __restrict__ Wpk,
                                                   const float* __restrict__ be,
                                                   const float* __restrict__ x,
                                                   const int* __restrict__ src,
                                                   const int* __restrict__ dst,
                                                   float* __restrict__ agg) {
  int e0   = blockIdx.x * 16;
  int w    = threadIdx.x >> 5;
  int lane = threadIdx.x & 31;
  int hi   = lane >> 4;
  int m    = lane & 15;
  const _Float16* atile = zembh + (size_t)blockIdx.x * 2048 + lane * 16;
  v8f c = {0.f, 0.f, 0.f, 0.f, 0.f, 0.f, 0.f, 0.f};
#pragma unroll
  for (int kk = 0; kk < 4; ++kk) {
    v16h a = *(const v16h*)(atile + kk * 512);
    v16h b = *(const v16h*)(Wpk + (((w * 4 + kk) * 32 + lane) << 4));
    c = __builtin_amdgcn_wmma_f32_16x16x32_f16(false, a, false, b, (short)0, c,
                                               false, false);
  }
  int col = w * 16 + m;
  float bias = be[col];
#pragma unroll
  for (int r = 0; r < 8; ++r) {
    int e = e0 + hi * 8 + r;                 // C/D layout: M = hi*8 + r
    int s = src[e], d = dst[e];
    float msg = c[r] + bias + x[(size_t)s * HID + col];
    msg = fmaxf(msg, 0.f);
    atomicAdd(&agg[(size_t)d * HID + col], msg);
  }
}

// ---------------------------------------------------------------------------
// out = elu((A0 [+ A1]) @ W + bias), M x 128 by 128 x 128, WMMA f32 accum.
// ---------------------------------------------------------------------------
__global__ __launch_bounds__(256) void k_gemm_bias_elu(const float* __restrict__ A0,
                                                       const float* __restrict__ A1,
                                                       const _Float16* __restrict__ Wpk,
                                                       const float* __restrict__ bias,
                                                       float* __restrict__ out) {
  int r0   = blockIdx.x * 16;
  int w    = threadIdx.x >> 5;
  int lane = threadIdx.x & 31;
  int hi   = lane >> 4;
  int m    = lane & 15;
  v8f c = {0.f, 0.f, 0.f, 0.f, 0.f, 0.f, 0.f, 0.f};
#pragma unroll
  for (int kk = 0; kk < 4; ++kk) {
    v16h a = load_a_frag(A0, A1, r0 + m, kk, hi);
    v16h b = *(const v16h*)(Wpk + (((w * 4 + kk) * 32 + lane) << 4));
    c = __builtin_amdgcn_wmma_f32_16x16x32_f16(false, a, false, b, (short)0, c,
                                               false, false);
  }
  int col = w * 16 + m;
  float b0 = bias[col];
#pragma unroll
  for (int r = 0; r < 8; ++r) {
    int row = r0 + hi * 8 + r;
    float v = c[r] + b0;
    v = (v > 0.f) ? v : (expf(v) - 1.f);     // ELU
    out[(size_t)row * HID + col] = v;
  }
}

// ---------------------------------------------------------------------------
// conv1 edge: ea = dot(z_emb[e,:], We[:,0]) + be; msg=relu(1+ea);
// scatter-add scalar into agg1[dst[e]]. One wave32 per edge; reads the
// packed-f16 z_emb layout (k = t*32 + lane -> jin = lane).
// ---------------------------------------------------------------------------
__global__ __launch_bounds__(256) void k_conv1_edge(const _Float16* __restrict__ zembh,
                                                    const float* __restrict__ We,
                                                    const float* __restrict__ be,
                                                    const int* __restrict__ dst,
                                                    float* __restrict__ agg1) {
  int e    = blockIdx.x * 8 + (threadIdx.x >> 5);
  int lane = threadIdx.x & 31;
  int fhi  = (lane >> 3) & 1;
  int fh   = (lane & 7) | ((lane >> 4) << 3);
  int plane = fhi * 16 + (e & 15);
  size_t base = (size_t)(e >> 4) * 2048 + plane * 16 + fh;
  float acc = 0.f;
#pragma unroll
  for (int t = 0; t < 4; ++t) {
    int k = t * 32 + lane;
    acc += (float)zembh[base + t * 512] * We[k];
  }
#pragma unroll
  for (int off = 16; off > 0; off >>= 1) acc += __shfl_xor(acc, off, 32);
  if (lane == 0) {
    float msg = fmaxf(1.f + acc + be[0], 0.f);
    atomicAdd(&agg1[dst[e]], msg);
  }
}

// conv1 node: h[n,j] = elu((1 + agg1[n]) * W0[j] + b0[j])   (rank-1 expand)
__global__ __launch_bounds__(256) void k_conv1_node(const float* __restrict__ agg1,
                                                    const float* __restrict__ W0,
                                                    const float* __restrict__ b0,
                                                    float* __restrict__ h) {
  int i = blockIdx.x * 256 + threadIdx.x;
  int n = i >> 7, j = i & 127;
  float t = 1.f + agg1[n];
  float v = t * W0[j] + b0[j];
  h[i] = (v > 0.f) ? v : (expf(v) - 1.f);
}

// Deterministic graph pooling (batch is sorted).
__global__ __launch_bounds__(128) void k_pool(const float* __restrict__ x,
                                              const int* __restrict__ batch,
                                              float* __restrict__ pooled) {
  int g = blockIdx.x, j = threadIdx.x;
  int lo = 0, hi = N_NODES;
  while (lo < hi) { int mid = (lo + hi) >> 1; if (batch[mid] < g) lo = mid + 1; else hi = mid; }
  float acc = 0.f;
  for (int n = lo; n < N_NODES && batch[n] == g; ++n) acc += x[(size_t)n * HID + j];
  pooled[g * HID + j] = acc;
}

// Final head: out[g,j] = hg[g,:] . lin2_W[:,j] + lin2_b[j], j < 10.
__global__ __launch_bounds__(256) void k_lin2(const float* __restrict__ hg,
                                              const float* __restrict__ W,
                                              const float* __restrict__ b,
                                              float* __restrict__ out) {
  int i = blockIdx.x * 256 + threadIdx.x;
  if (i >= N_GRAPHS * 10) return;
  int g = i / 10, j = i % 10;
  float acc = b[j];
#pragma unroll 4
  for (int k = 0; k < HID; ++k) acc += hg[g * HID + k] * W[k * 10 + j];
  out[i] = acc;
}

// ---------------------------------------------------------------------------
extern "C" void kernel_launch(void* const* d_in, const int* in_sizes, int n_in,
                              void* d_out, int out_size, void* d_ws, size_t ws_size,
                              hipStream_t stream) {
  (void)in_sizes; (void)n_in; (void)out_size; (void)ws_size;

  const int*   edge_index = (const int*)d_in[0];
  const int*   src        = edge_index;
  const int*   dst        = edge_index + N_EDGES;
  const int*   batch      = (const int*)d_in[1];
  const int*   pos_index  = (const int*)d_in[2];
  const float* pos_enc    = (const float*)d_in[3];
  const int*   pos_batch  = (const int*)d_in[4];
  const float* zw         = (const float*)d_in[5];
  const float* c1_W0      = (const float*)d_in[6];
  const float* c1_b0      = (const float*)d_in[7];
  const float* c1_W1      = (const float*)d_in[8];
  const float* c1_b1      = (const float*)d_in[9];
  const float* c1_We      = (const float*)d_in[10];
  const float* c1_be      = (const float*)d_in[11];
  const float* cs_W0      = (const float*)d_in[12];
  const float* cs_b0      = (const float*)d_in[13];
  const float* cs_W1      = (const float*)d_in[14];
  const float* cs_b1      = (const float*)d_in[15];
  const float* cs_We      = (const float*)d_in[16];
  const float* cs_be      = (const float*)d_in[17];
  const float* lin1_W     = (const float*)d_in[18];
  const float* lin1_b     = (const float*)d_in[19];
  const float* lin2_W     = (const float*)d_in[20];
  const float* lin2_b     = (const float*)d_in[21];
  float* out = (float*)d_out;

  char* ws = (char*)d_ws;
  size_t off = 0;
  auto carve = [&](size_t bytes) -> char* {
    char* p = ws + off;
    off += (bytes + 255) & ~(size_t)255;
    return p;
  };
  _Float16* zembh  = (_Float16*)carve((size_t)N_EDGES * HID * sizeof(_Float16)); // 154 MB
  float*    agg    = (float*)carve((size_t)N_NODES * HID * sizeof(float));
  float*    hbuf   = (float*)carve((size_t)N_NODES * HID * sizeof(float));
  float*    xbuf   = (float*)carve((size_t)N_NODES * HID * sizeof(float));
  float*    agg1   = (float*)carve((size_t)N_NODES * sizeof(float));
  float*    pooled = (float*)carve((size_t)N_GRAPHS * HID * sizeof(float));
  float*    hg     = (float*)carve((size_t)N_GRAPHS * HID * sizeof(float));
  _Float16* Wepk   = (_Float16*)carve((size_t)HID * HID * sizeof(_Float16));
  _Float16* W0pk   = (_Float16*)carve((size_t)HID * HID * sizeof(_Float16));
  _Float16* W1pk   = (_Float16*)carve((size_t)HID * HID * sizeof(_Float16));

  // 1. edge positional embedding (deterministic segmented sum, packed f16 out)
  k_zemb<<<N_EDGES, 128, 0, stream>>>(zw, pos_index, pos_enc, pos_batch, zembh);

  // 2. conv1 (edge attr is H->1, node features start as ones[N,1])
  hipMemsetAsync(agg1, 0, (size_t)N_NODES * sizeof(float), stream);
  k_conv1_edge<<<N_EDGES / 8, 256, 0, stream>>>(zembh, c1_We, c1_be, dst, agg1);
  k_conv1_node<<<(N_NODES * HID) / 256, 256, 0, stream>>>(agg1, c1_W0, c1_b0, hbuf);
  k_pack_w<<<64, 256, 0, stream>>>(c1_W1, W1pk);
  k_gemm_bias_elu<<<N_NODES / 16, 256, 0, stream>>>(hbuf, nullptr, W1pk, c1_b1, xbuf);

  // 3. conv layers 2..4
  for (int l = 0; l < 3; ++l) {
    k_pack_w<<<64, 256, 0, stream>>>(cs_We + (size_t)l * HID * HID, Wepk);
    k_pack_w<<<64, 256, 0, stream>>>(cs_W0 + (size_t)l * HID * HID, W0pk);
    k_pack_w<<<64, 256, 0, stream>>>(cs_W1 + (size_t)l * HID * HID, W1pk);
    hipMemsetAsync(agg, 0, (size_t)N_NODES * HID * sizeof(float), stream);
    k_edge_gine<<<N_EDGES / 16, 256, 0, stream>>>(zembh, Wepk, cs_be + l * HID,
                                                  xbuf, src, dst, agg);
    k_gemm_bias_elu<<<N_NODES / 16, 256, 0, stream>>>(xbuf, agg, W0pk,
                                                      cs_b0 + l * HID, hbuf);
    k_gemm_bias_elu<<<N_NODES / 16, 256, 0, stream>>>(hbuf, nullptr, W1pk,
                                                      cs_b1 + l * HID, xbuf);
  }

  // 4. pool + MLP head
  k_pool<<<N_GRAPHS, 128, 0, stream>>>(xbuf, batch, pooled);
  k_pack_w<<<64, 256, 0, stream>>>(lin1_W, W0pk);
  k_gemm_bias_elu<<<N_GRAPHS / 16, 256, 0, stream>>>(pooled, nullptr, W0pk, lin1_b, hg);
  k_lin2<<<(N_GRAPHS * 10 + 255) / 256, 256, 0, stream>>>(hg, lin2_W, lin2_b, out);
}
